// TS_Corr_75703093559348
// MI455X (gfx1250) — compile-verified, matrix operand order checked by reading, not run on hardware
//
#include <hip/hip_runtime.h>

typedef __attribute__((ext_vector_type(16))) _Float16     v16h;
typedef __attribute__((ext_vector_type(8)))  float        v8f;
typedef __attribute__((ext_vector_type(4)))  unsigned int v4u;
typedef __attribute__((ext_vector_type(8)))  int          v8i;
typedef __attribute__((ext_vector_type(4)))  int          v4i;

#define DAYS   32
#define NCH    64
#define TLEN   2048
#define BATCH  8
#define TWIN   2017        // (2048 - 32) / 1 + 1
#define NPAIR  2016        // 64*63/2
#define WPB    8           // waves per block
#define SROW   2032        // LDS row stride: NPAIR + trash slot + pad
#define EPS31  3.1e-5f     // 31 * 1e-6

// One wave32 per (b, t) window.
// Gram matrix cent^T * cent via split-f16 WMMA (hi*hi + hi*lo + lo*hi) for
// ~fp32 accuracy. Output phase is the roofline (130 MB @ 23.3 TB/s): stage
// the packed triangle in LDS, then one TDM tensor_store_from_lds DMAs the
// contiguous 8064 B block to HBM (full-line writes, wave retires on TENSORcnt).
__global__ __launch_bounds__(256) void ts_corr_wmma_kernel(
    const float* __restrict__ x, float* __restrict__ out) {
  __shared__ float stage[WPB][SROW];

  const int wslot = threadIdx.x >> 5;
  // Wave-uniform window id in an SGPR: scalar bases, saddr-form loads.
  const int wid = __builtin_amdgcn_readfirstlane(blockIdx.x * WPB + wslot);
  if (wid >= BATCH * TWIN) return;            // uniform guard (exact grid)

  const int lane  = threadIdx.x & 31;
  const int half  = lane >> 4;                // which K-half this lane packs
  const int m16   = lane & 15;                // row/col within 16-wide block
  const int kbase = half * 8;

  const int b = wid / TWIN;                   // SALU
  const int t = wid - b * TWIN;
  const float* __restrict__ xw = x + ((size_t)b * TLEN + t) * NCH;

  // One per-lane element offset shared by all 64 loads; (e, cb) part is a
  // compile-time constant folded into the 24-bit instruction offset.
  const int voff = kbase * NCH + m16;

  v16h  ahi[4], alo[4];
  float stdv[4];

  // Centered, split-precision A/B tiles per 16-channel block. Each lane loads
  // only the 16 K-values its half packs (CDNA5 16-bit A-matrix 16x32 layout:
  // e -> K = (e<8 ? e : e+8) + 8*half); full-window mean/var via xor-shuffle.
  #pragma unroll
  for (int cb = 0; cb < 4; ++cb) {
    float w[16];
    #pragma unroll
    for (int e = 0; e < 16; ++e) {
      const int ep = (e < 8) ? e : e + 8;
      w[e] = xw[voff + (ep * NCH + cb * 16)];
    }

    float s = 0.f;
    #pragma unroll
    for (int e = 0; e < 16; ++e) s += w[e];
    s += __shfl_xor(s, 16, 32);
    const float mean = s * (1.0f / DAYS);

    float ss = 0.f;
    #pragma unroll
    for (int e = 0; e < 16; ++e) { w[e] -= mean; ss += w[e] * w[e]; }
    ss += __shfl_xor(ss, 16, 32);
    stdv[cb] = __builtin_amdgcn_sqrtf(ss * (1.0f / (DAYS - 1)));  // ddof=1

    v16h hi, lo;
    #pragma unroll
    for (int e = 0; e < 16; ++e) {
      const float    c0 = w[e];
      const _Float16 h  = (_Float16)c0;
      hi[e] = h;
      lo[e] = (_Float16)(c0 - (float)h);
    }
    ahi[cb] = hi;
    alo[cb] = lo;
  }

  // Fold the 1/31 into the denominator: (G/31)/(si*sj+eps) = G/(31*si*sj+31eps)
  float std31[4];
  #pragma unroll
  for (int cb = 0; cb < 4; ++cb) std31[cb] = stdv[cb] * (float)(DAYS - 1);

  float* __restrict__ sw = &stage[wslot][0];

  // Upper-triangle 16x16 tiles of the 64x64 correlation matrix -> LDS stage.
  #pragma unroll
  for (int ti = 0; ti < 4; ++ti) {
    float stdTi[8];
    int   rowOff[8];
    #pragma unroll
    for (int r = 0; r < 8; ++r) {
      stdTi[r] = __shfl(stdv[ti], r + kbase, 32);   // C/D layout M = r+8*half
      const int i = ti * 16 + r + kbase;
      rowOff[r] = (i * (2 * NCH - 1 - i)) / 2 - i - 1;  // triu packing base
    }

    #pragma unroll
    for (int tj = ti; tj < 4; ++tj) {
      v8f acc = {};
      // split-precision Gram: hi*hi + hi*lo + lo*hi  (lo*lo negligible)
      acc = __builtin_amdgcn_wmma_f32_16x16x32_f16(
          false, ahi[ti], false, ahi[tj], (short)0, acc, false, false);
      acc = __builtin_amdgcn_wmma_f32_16x16x32_f16(
          false, ahi[ti], false, alo[tj], (short)0, acc, false, false);
      acc = __builtin_amdgcn_wmma_f32_16x16x32_f16(
          false, alo[ti], false, ahi[tj], (short)0, acc, false, false);

      const int   j   = tj * 16 + m16;         // C/D layout: N = lane%16
      const float s31 = std31[tj];

      #pragma unroll
      for (int r = 0; r < 8; ++r) {
        const float corr = acc[r] *
            __builtin_amdgcn_rcpf(stdTi[r] * s31 + EPS31);
        int idx = rowOff[r] + j;
        if (ti == tj)                           // compile-time tile class:
          idx = (m16 > r + kbase) ? idx : NPAIR;  // j<=i -> LDS trash slot
        sw[idx] = corr;                         // unconditional ds_store
      }
    }
  }

  // Tensor ops are unordered vs DS: drain LDS writes before TDM issue.
  asm volatile("s_wait_dscnt 0x0" ::: "memory");

  // ---- Tensor DMA descriptor: 1D tile, 2016 x 4B, LDS -> global ----
  const unsigned lds_byte =
      (unsigned)__builtin_amdgcn_readfirstlane((int)(unsigned)(size_t)sw);
  const unsigned long long ga =
      (unsigned long long)(size_t)(out + (size_t)wid * NPAIR);

  v4u g0;
  g0.x = 1u;                                   // count=1, is_restore=0
  g0.y = lds_byte;                             // lds_addr
  g0.z = (unsigned)ga;                         // global_addr[31:0]
  g0.w = ((unsigned)(ga >> 32) & 0x01FFFFFFu)  // global_addr[56:32]
         | 0x80000000u;                        // type=2 ("image")

  v8i g1;
  g1[0] = 0x00020000;                          // data_size=2 (4 bytes)
  g1[1] = (int)((unsigned)NPAIR << 16);        // tensor_dim0[15:0] @ bits63:48
  g1[2] = (int)(1u << 16);                     // tensor_dim0 hi=0 | tensor_dim1=1
  g1[3] = (int)((unsigned)NPAIR << 16);        // tensor_dim1 hi=0 | tile_dim0=2016
  g1[4] = 1;                                   // tile_dim1=1, tile_dim2=0
  g1[5] = NPAIR;                               // tensor_dim0_stride[31:0]
  g1[6] = 0;                                   // stride0 hi, stride1 lo (unused)
  g1[7] = 0;

  const v4i gz4 = {0, 0, 0, 0};                // groups 2/3 unused (<=2D)
  const v8i gz8 = {0, 0, 0, 0, 0, 0, 0, 0};    // 6-arg form: extra group
  __builtin_amdgcn_tensor_store_from_lds(g0, g1, gz4, gz4, gz8, 0);

  // Keep LDS alive until the TDM has consumed it.
  __builtin_amdgcn_s_wait_tensorcnt(0);
}

extern "C" void kernel_launch(void* const* d_in, const int* in_sizes, int n_in,
                              void* d_out, int out_size, void* d_ws, size_t ws_size,
                              hipStream_t stream) {
  const float* x   = (const float*)d_in[0];
  float*       out = (float*)d_out;

  const int nwaves = BATCH * TWIN;                       // 16136
  const int blocks = (nwaves + WPB - 1) / WPB;           // 2017

  hipLaunchKernelGGL(ts_corr_wmma_kernel, dim3(blocks), dim3(WPB * 32),
                     0, stream, x, out);
}
